// RTRBM_20375324852725
// MI455X (gfx1250) — matrix operand order repaired and checked by reading, not compiled, not run on hardware
//
#include <hip/hip_runtime.h>
#include <stdint.h>

typedef __attribute__((ext_vector_type(16))) _Float16 v16h;
typedef __attribute__((ext_vector_type(8)))  _Float16 v8h;
typedef __attribute__((ext_vector_type(8)))  float    v8f;

#define N_VIS   784
#define N_HID   500
#define T_STEPS 256
#define BATCH   256
#define NTB     (T_STEPS * BATCH)

// padded GEMM dims
#define HID_P   512   // hidden M pad / visible & U K pad (16-tile multiples)
#define VIS_PK  800   // K pad for W in hidden gemm (25 * 32)
#define VIS_PM  896   // M pad for visible gemm (7 * 128)

#define BM      128   // block M tile (8 waves * 16)
#define BN      64    // block N tile (4 wmma tiles per wave)
#define LDB     40    // LDS leading dim (halves): 80B row -> conflict-free frags

// ---------------------------------------------------------------------------
__device__ __forceinline__ float u01(uint32_t x) {
    x ^= x >> 17; x *= 0xed5ad4bbu;
    x ^= x >> 11; x *= 0xac4c1b51u;
    x ^= x >> 15; x *= 0x31848babu;
    x ^= x >> 14;
    return (float)(x >> 8) * (1.0f / 16777216.0f);
}

__device__ __forceinline__ float sigmoidf(float x) {
    return 1.0f / (1.0f + __expf(-x));
}

// ---------------------------------------------------------------------------
// A fragment (16x32 f16 row-major): lanes 0-15 M=lane, K k0..k0+7 / k0+16..23;
// lanes 16-31 same rows, K offset +8 (ISA 7.12.2 16-bit A layout).
__device__ __forceinline__ v16h load_A_frag(const _Float16* A, int lda,
                                            int row, int kTile) {
    int lane = threadIdx.x & 31;
    int k0 = kTile + ((lane >> 4) << 3);
    const _Float16* p = A + (size_t)row * lda + k0;
    union { v16h v; v8h h[2]; } u;
    u.h[0] = *(const v8h*)(p);
    u.h[1] = *(const v8h*)(p + 16);
    return u.v;
}

// B fragment from transposed LDS tile Bt[n*LDB + k]: lane's 16 halves are
// two 16B ds loads. Lanes 0-15: K 0-7 / 16-23; lanes 16-31: K 8-15 / 24-31.
__device__ __forceinline__ v16h load_B_frag(const _Float16* Bt, int nt) {
    int lane = threadIdx.x & 31;
    int n  = (lane & 15) + nt * 16;
    int kb = (lane >> 4) << 3;
    const _Float16* p = Bt + n * LDB + kb;
    union { v16h v; v8h h[2]; } u;
    u.h[0] = *(const v8h*)(p);
    u.h[1] = *(const v8h*)(p + 16);
    return u.v;
}

// Stage a 32 x BN B-tile into transposed LDS. 256 threads, 8 elems each,
// packed as 32-bit LDS stores. GUARD path is branch-free (clamp + select).
template <bool GUARD, typename SrcT>
__device__ __forceinline__ void stage_tile(_Float16* Bt, const SrcT* src,
                                           size_t ld, int kBase, int kReal,
                                           int nBase) {
    int tid = threadIdx.x;
    int n  = tid & (BN - 1);
    int rb = (tid >> 6) * 8;
    const SrcT* col = src + nBase + n;
#pragma unroll
    for (int j = 0; j < 4; ++j) {
        int r  = rb + 2 * j;
        int k0 = kBase + r;
        float f0, f1;
        if (GUARD) {
            int kc0 = (k0     < kReal) ? k0     : 0;
            int kc1 = (k0 + 1 < kReal) ? k0 + 1 : 0;
            f0 = (float)col[(size_t)kc0 * ld];
            f1 = (float)col[(size_t)kc1 * ld];
            f0 = (k0     < kReal) ? f0 : 0.0f;
            f1 = (k0 + 1 < kReal) ? f1 : 0.0f;
        } else {
            f0 = (float)col[(size_t)k0 * ld];
            f1 = (float)col[(size_t)(k0 + 1) * ld];
        }
        union { _Float16 h[2]; uint32_t u; } pk;
        pk.h[0] = (_Float16)f0; pk.h[1] = (_Float16)f1;
        *(uint32_t*)(Bt + n * LDB + r) = pk.u;
    }
}

// ---------------------------------------------------------------------------
// Prep: f32 -> f16 padded matrices, zero r_lag
// ---------------------------------------------------------------------------
__global__ __launch_bounds__(256) void rtrbm_prep(
    const float* __restrict__ W, const float* __restrict__ U,
    _Float16* Wh, _Float16* Wth, _Float16* Uh,
    _Float16* rlag0, _Float16* hbuf) {
    size_t tid    = (size_t)blockIdx.x * blockDim.x + threadIdx.x;
    size_t stride = (size_t)gridDim.x * blockDim.x;
    for (size_t i = tid; i < (size_t)HID_P * VIS_PK; i += stride) {
        int m = (int)(i / VIS_PK), k = (int)(i % VIS_PK);
        Wh[i] = (m < N_HID && k < N_VIS) ? (_Float16)W[(size_t)m * N_VIS + k]
                                         : (_Float16)0.0f;
    }
    for (size_t i = tid; i < (size_t)VIS_PM * HID_P; i += stride) {
        int m = (int)(i / HID_P), k = (int)(i % HID_P);
        Wth[i] = (m < N_VIS && k < N_HID) ? (_Float16)W[(size_t)k * N_VIS + m]
                                          : (_Float16)0.0f;
    }
    for (size_t i = tid; i < (size_t)HID_P * HID_P; i += stride) {
        int m = (int)(i / HID_P), k = (int)(i % HID_P);
        Uh[i] = (m < N_HID && k < N_HID) ? (_Float16)U[(size_t)m * N_HID + k]
                                         : (_Float16)0.0f;
    }
    for (size_t i = tid; i < (size_t)HID_P * BATCH; i += stride) {
        rlag0[i] = (_Float16)0.0f;
        hbuf[i]  = (_Float16)0.0f;
    }
}

// ---------------------------------------------------------------------------
// Time-independent precompute: WV = W @ v  (512 x 65536, K=800), f16 out.
// grid (4, 1024) -- fills the chip; v read exactly once here.
// ---------------------------------------------------------------------------
__global__ __launch_bounds__(256) void rtrbm_precompute(
    const float* __restrict__ v, const _Float16* __restrict__ Wh,
    _Float16* __restrict__ WVh) {
    __shared__ _Float16 Bt[3][BN * LDB];
    const int NT = VIS_PK / 32;                       // 25
    int wave = threadIdx.x >> 5, lane = threadIdx.x & 31;
    int mBase   = blockIdx.x * BM + wave * 16;
    int colBase = blockIdx.y * BN;
    v8f c[4] = {};

    auto stage_idx = [&](int idx, int buf) {
        if (idx == NT - 1)
            stage_tile<true >(Bt[buf], v, (size_t)NTB, idx * 32, N_VIS, colBase);
        else
            stage_tile<false>(Bt[buf], v, (size_t)NTB, idx * 32, N_VIS, colBase);
    };
    stage_idx(0, 0);
    stage_idx(1, 1);
    for (int i = 0; i < NT; ++i) {
        __syncthreads();
        v16h a = load_A_frag(Wh, VIS_PK, mBase + (lane & 15), i * 32);
#pragma unroll
        for (int nt = 0; nt < 4; ++nt) {
            v16h b = load_B_frag(Bt[i % 3], nt);
            c[nt] = __builtin_amdgcn_wmma_f32_16x16x32_f16(
                false, a, false, b, (short)0, c[nt], false, false);
        }
        if (i + 2 < NT) stage_idx(i + 2, (i + 2) % 3);
    }
    int n = lane & 15, mh = (lane >> 4) & 1;
#pragma unroll
    for (int nt = 0; nt < 4; ++nt) {
        int cg = colBase + nt * 16 + n;
#pragma unroll
        for (int i = 0; i < 8; ++i) {
            int m = mBase + i + mh * 8;
            if (m < N_HID) WVh[(size_t)m * NTB + cg] = (_Float16)c[nt][i];
        }
    }
}

// ---------------------------------------------------------------------------
// Hidden step. USE_WV: pre = WV[t] + U@r_lag + bias (16 K-tiles).
//              else  : pre = W@v_t + U@r_lag + bias (41 K-tiles, fused).
// grid (4, 4), 256 threads; block tile 128x64, wave tile 16x64.
// ---------------------------------------------------------------------------
template <bool USE_WV>
__global__ __launch_bounds__(256) void rtrbm_hidden(
    const float* __restrict__ v, const _Float16* __restrict__ WVh,
    const _Float16* __restrict__ Wh, const _Float16* __restrict__ Uh,
    const _Float16* __restrict__ rlag_in,
    const float* __restrict__ b_h, const float* __restrict__ b_init,
    _Float16* __restrict__ rlag_out, _Float16* __restrict__ hbuf,
    float* __restrict__ outR, int t) {
    __shared__ _Float16 Bt[3][BN * LDB];
    const int NT_W = USE_WV ? 0 : (VIS_PK / 32);      // 0 or 25
    const int NT   = NT_W + HID_P / 32;               // 16 or 41
    int wave = threadIdx.x >> 5, lane = threadIdx.x & 31;
    int mBase = blockIdx.x * BM + wave * 16;
    int nBase = blockIdx.y * BN;
    const float* vsrc = v + (size_t)t * BATCH;
    v8f c[4] = {};

    auto stage_idx = [&](int idx, int buf) {
        if (!USE_WV && idx < NT_W) {
            if (idx == NT_W - 1)
                stage_tile<true >(Bt[buf], vsrc, (size_t)NTB, idx * 32, N_VIS, nBase);
            else
                stage_tile<false>(Bt[buf], vsrc, (size_t)NTB, idx * 32, N_VIS, nBase);
        } else {
            stage_tile<false>(Bt[buf], rlag_in, (size_t)BATCH,
                              (idx - NT_W) * 32, HID_P, nBase);
        }
    };
    stage_idx(0, 0);
    stage_idx(1, 1);
    for (int i = 0; i < NT; ++i) {
        __syncthreads();
        const _Float16* A; int lda, kt;
        if (!USE_WV && i < NT_W) { A = Wh; lda = VIS_PK; kt = i * 32; }
        else                     { A = Uh; lda = HID_P;  kt = (i - NT_W) * 32; }
        v16h a = load_A_frag(A, lda, mBase + (lane & 15), kt);
#pragma unroll
        for (int nt = 0; nt < 4; ++nt) {
            v16h b = load_B_frag(Bt[i % 3], nt);
            c[nt] = __builtin_amdgcn_wmma_f32_16x16x32_f16(
                false, a, false, b, (short)0, c[nt], false, false);
        }
        if (i + 2 < NT) stage_idx(i + 2, (i + 2) % 3);
    }

    int n = lane & 15, mh = (lane >> 4) & 1;
    const float* bias = (t == 0) ? b_init : b_h;
#pragma unroll
    for (int nt = 0; nt < 4; ++nt) {
        int ng = nBase + nt * 16 + n;
#pragma unroll
        for (int i = 0; i < 8; ++i) {
            int   m = mBase + i + mh * 8;
            float r = 0.0f, h = 0.0f;
            if (m < N_HID) {
                float pre = c[nt][i] + bias[m];
                if (USE_WV)
                    pre += (float)WVh[(size_t)m * NTB + (size_t)t * BATCH + ng];
                r = sigmoidf(pre);
                uint32_t key = ((uint32_t)(t * HID_P + m) * 0x9E3779B9u)
                             ^ ((uint32_t)ng * 0x85EBCA6Bu) ^ 0xA511E9B3u;
                h = (u01(key) < r) ? 1.0f : 0.0f;
                outR[(size_t)m * NTB + (size_t)t * BATCH + ng] = h;
            }
            rlag_out[(size_t)m * BATCH + ng] = (_Float16)r;
            hbuf[(size_t)m * BATCH + ng]     = (_Float16)h;
        }
    }
}

// ---------------------------------------------------------------------------
// Visible step: v_model = bernoulli(sigmoid(W.T @ h + b_v)). grid (7, 4).
// ---------------------------------------------------------------------------
__global__ __launch_bounds__(256) void rtrbm_visible(
    const _Float16* __restrict__ Wth, const _Float16* __restrict__ hbuf,
    const float* __restrict__ b_v, float* __restrict__ outV, int t) {
    __shared__ _Float16 Bt[3][BN * LDB];
    const int NT = HID_P / 32;                        // 16
    int wave = threadIdx.x >> 5, lane = threadIdx.x & 31;
    int mBase = blockIdx.x * BM + wave * 16;
    int nBase = blockIdx.y * BN;
    v8f c[4] = {};

    stage_tile<false>(Bt[0], hbuf, (size_t)BATCH, 0,  HID_P, nBase);
    stage_tile<false>(Bt[1], hbuf, (size_t)BATCH, 32, HID_P, nBase);
    for (int i = 0; i < NT; ++i) {
        __syncthreads();
        v16h a = load_A_frag(Wth, HID_P, mBase + (lane & 15), i * 32);
#pragma unroll
        for (int nt = 0; nt < 4; ++nt) {
            v16h b = load_B_frag(Bt[i % 3], nt);
            c[nt] = __builtin_amdgcn_wmma_f32_16x16x32_f16(
                false, a, false, b, (short)0, c[nt], false, false);
        }
        if (i + 2 < NT)
            stage_tile<false>(Bt[(i + 2) % 3], hbuf, (size_t)BATCH,
                              (i + 2) * 32, HID_P, nBase);
    }

    int n = lane & 15, mh = (lane >> 4) & 1;
#pragma unroll
    for (int nt = 0; nt < 4; ++nt) {
        int ng = nBase + nt * 16 + n;
#pragma unroll
        for (int i = 0; i < 8; ++i) {
            int m = mBase + i + mh * 8;
            if (m < N_VIS) {
                float vm = sigmoidf(c[nt][i] + b_v[m]);
                uint32_t key = ((uint32_t)(t * 1024 + m) * 0xC2B2AE35u)
                             ^ ((uint32_t)ng * 0x27D4EB2Fu) ^ 0x5BD1E995u;
                outV[(size_t)m * NTB + (size_t)t * BATCH + ng] =
                    (u01(key) < vm) ? 1.0f : 0.0f;
            }
        }
    }
}

// ---------------------------------------------------------------------------
extern "C" void kernel_launch(void* const* d_in, const int* in_sizes, int n_in,
                              void* d_out, int out_size, void* d_ws,
                              size_t ws_size, hipStream_t stream) {
    const float* v      = (const float*)d_in[0];
    const float* W      = (const float*)d_in[1];
    const float* U      = (const float*)d_in[2];
    const float* b_h    = (const float*)d_in[3];
    const float* b_v    = (const float*)d_in[4];
    const float* b_init = (const float*)d_in[5];

    float* outV = (float*)d_out;
    float* outR = outV + (size_t)N_VIS * NTB;

    _Float16* p     = (_Float16*)d_ws;
    _Float16* Wh    = p; p += (size_t)HID_P * VIS_PK;
    _Float16* Wth   = p; p += (size_t)VIS_PM * HID_P;
    _Float16* Uh    = p; p += (size_t)HID_P * HID_P;
    _Float16* rlagA = p; p += (size_t)HID_P * BATCH;
    _Float16* rlagB = p; p += (size_t)HID_P * BATCH;
    _Float16* hbuf  = p; p += (size_t)HID_P * BATCH;
    _Float16* WVh   = p; p += (size_t)HID_P * NTB;     // 64 MB, optional

    size_t need_big = (size_t)((char*)p - (char*)d_ws);
    bool   use_wv   = ws_size >= need_big;

    rtrbm_prep<<<512, 256, 0, stream>>>(W, U, Wh, Wth, Uh, rlagA, hbuf);
    if (use_wv)
        rtrbm_precompute<<<dim3(HID_P / BM, NTB / BN), 256, 0, stream>>>(v, Wh, WVh);

    dim3 gh(HID_P / BM, BATCH / BN);    // (4, 4)
    dim3 gv(VIS_PM / BM, BATCH / BN);   // (7, 4)
    for (int t = 0; t < T_STEPS; ++t) {
        const _Float16* rin  = (t & 1) ? rlagB : rlagA;
        _Float16*       rout = (t & 1) ? rlagA : rlagB;
        if (use_wv)
            rtrbm_hidden<true ><<<gh, 256, 0, stream>>>(v, WVh, Wh, Uh, rin,
                                                        b_h, b_init, rout,
                                                        hbuf, outR, t);
        else
            rtrbm_hidden<false><<<gh, 256, 0, stream>>>(v, WVh, Wh, Uh, rin,
                                                        b_h, b_init, rout,
                                                        hbuf, outR, t);
        rtrbm_visible<<<gv, 256, 0, stream>>>(Wth, hbuf, b_v, outV, t);
    }
}
